// MyLSTM_4741643895182
// MI455X (gfx1250) — compile-verified, matrix operand order checked by reading, not run on hardware
//
#include <hip/hip_runtime.h>
#include <math.h>

// ---------------------------------------------------------------------------
// 2-layer LSTM (H=128, SEQ=2048, BATCH=256) on gfx1250 (MI455X).
// 16 workgroups x 256 threads (8 wave32). Each WG owns 16 batch rows and runs
// the full recurrence for both layers. Wave w owns hidden columns [16w,16w+16)
// => its 4 gate tiles (i,f,g,o) and its slice of cell state stay in registers.
// All weight WMMA B-fragments are register-resident (48 x v16h per wave).
// h vectors are shared through LDS (f16, padded stride 136 halves).
// Gate nonlinearities use hardware V_TANH_F32 (branch-free), sigmoid via the
// tanh identity. x loads are software-pipelined one step ahead.
// ---------------------------------------------------------------------------

typedef _Float16 v16h __attribute__((ext_vector_type(16)));
typedef _Float16 v8h  __attribute__((ext_vector_type(8)));
typedef float    v8f  __attribute__((ext_vector_type(8)));

#define HID    128
#define BATCH  256
#define SEQN   2048
#define ROWS   16      // batch rows per workgroup
#define NWG    (BATCH / ROWS)
#define STR    136     // padded LDS row stride in halves (272B: 16B aligned, bank-rotated)

__device__ __forceinline__ int kmap(int j, int lh) {
    // 16-bit WMMA fragment K map (ISA 7.12.2): regs 0-3 hold K = 2j + 8*half,
    // regs 4-7 hold K = 16 + 2(j-4) + 8*half (each reg packs k, k+1).
    return (j < 4) ? (2 * j + 8 * lh) : (16 + 2 * (j - 4) + 8 * lh);
}

// Branch-free tanh: prefer gfx1250 hardware V_TANH_F32.
__device__ __forceinline__ float fast_tanh(float x) {
#if __has_builtin(__builtin_amdgcn_tanhf)
    return __builtin_amdgcn_tanhf(x);
#else
    // 1 - 2/(e^{2x}+1): branch-free, correct limits at +/-inf.
    return 1.0f - 2.0f / (__expf(2.0f * x) + 1.0f);
#endif
}

// sigmoid(x) = 0.5*tanh(x/2) + 0.5  (single transcendental op, branch-free)
__device__ __forceinline__ float fast_sigm(float x) {
    return fmaf(0.5f, fast_tanh(0.5f * x), 0.5f);
}

// B-matrix (32x16 f16) fragment for gates = h @ W^T:  B(k,n) = W[nrow][k].
// lane: n = lane%16 -> weight row nrow; pack per kmap from row-major f32 W.
__device__ __forceinline__ v16h load_b_frag(const float* __restrict__ W,
                                            int nrow, int kbase, int lh) {
    v16h r;
#pragma unroll
    for (int j = 0; j < 8; ++j) {
        const float2 f = *(const float2*)(W + nrow * HID + kbase + kmap(j, lh));
        r[2 * j]     = (_Float16)f.x;
        r[2 * j + 1] = (_Float16)f.y;
    }
    return r;
}

// A-matrix (16x32 f16) fragment from an LDS h buffer: lane reads row m=lane%16;
// regs 0-3 are 8 consecutive halves at kbase+8*half, regs 4-7 at +16.
__device__ __forceinline__ v16h load_a_frag(const _Float16* __restrict__ hbuf,
                                            int kbase, int li, int lh) {
    const _Float16* p = hbuf + li * STR + kbase + 8 * lh;
    v8h lo = *(const v8h*)p;          // ds_load_b128
    v8h hi = *(const v8h*)(p + 16);   // ds_load_b128
    return __builtin_shufflevector(lo, hi, 0, 1, 2, 3, 4, 5, 6, 7,
                                           8, 9, 10, 11, 12, 13, 14, 15);
}

__global__ __launch_bounds__(256, 1)
void lstm2_wmma_kernel(const float* __restrict__ x,
                       const float* __restrict__ Wih0, const float* __restrict__ Whh0,
                       const float* __restrict__ bih0, const float* __restrict__ bhh0,
                       const float* __restrict__ Wih1, const float* __restrict__ Whh1,
                       const float* __restrict__ bih1, const float* __restrict__ bhh1,
                       const float* __restrict__ Wlin, const float* __restrict__ blin,
                       float* __restrict__ out) {
    __shared__ __align__(16) _Float16 lds_h0[ROWS * STR];  // layer-0 hidden (f16)
    __shared__ __align__(16) _Float16 lds_h1[ROWS * STR];  // layer-1 hidden (f16)

    const int tid  = threadIdx.x;
    const int w    = tid >> 5;        // wave 0..7 -> hidden cols [16w,16w+16)
    const int lane = tid & 31;
    const int li   = lane & 15;
    const int lh   = lane >> 4;
    const int rb   = blockIdx.x * ROWS;

    // ---- one-time: load register-resident weight fragments + per-lane scalars
    v16h w0[4][4];     // W_hh_l0  (K=128)
    v16h w1i[4][4];    // W_ih_l1  (K=128)
    v16h w1h[4][4];    // W_hh_l1  (K=128)
    float bias0[4], bias1[4], wx0[4];
#pragma unroll
    for (int g = 0; g < 4; ++g) {
        const int col = 128 * g + 16 * w + li;   // gate-matrix column / weight row
        bias0[g] = bih0[col] + bhh0[col];
        bias1[g] = bih1[col] + bhh1[col];
        wx0[g]   = Wih0[col];                    // INPUT_DIM==1: W_ih_l0 is (512,1)
#pragma unroll
        for (int kc = 0; kc < 4; ++kc) {
            w0[g][kc]  = load_b_frag(Whh0, col, 32 * kc, lh);
            w1i[g][kc] = load_b_frag(Wih1, col, 32 * kc, lh);
            w1h[g][kc] = load_b_frag(Whh1, col, 32 * kc, lh);
        }
    }

    // ---- zero initial h (LDS) and c (registers)
    for (int i = tid; i < ROWS * STR; i += 256) {
        lds_h0[i] = (_Float16)0.f;
        lds_h1[i] = (_Float16)0.f;
    }
    v8f c0 = {};   // layer-0 cell state: elem j <-> (row j+8*lh, col 16w+li)
    v8f c1 = {};   // layer-1 cell state

    // ---- prime the x pipeline for t=0
    float xv[8];
#pragma unroll
    for (int j = 0; j < 8; ++j)
        xv[j] = x[rb + j + 8 * lh];

    __syncthreads();

    for (int t = 0; t < SEQN; ++t) {
        // ================= layer 0 =================
        // gates0 = x_t * wih0 + h0 @ Whh0^T + b0   (x contribution is an outer
        // product since INPUT_DIM==1: folded into the accumulator init).
        v8f acc[4];
#pragma unroll
        for (int g = 0; g < 4; ++g)
#pragma unroll
            for (int j = 0; j < 8; ++j)
                acc[g][j] = fmaf(xv[j], wx0[g], bias0[g]);

        // batch all A-fragment DS loads, then run the WMMA chain
        {
            v16h a0[4];
#pragma unroll
            for (int kc = 0; kc < 4; ++kc)
                a0[kc] = load_a_frag(lds_h0, 32 * kc, li, lh);
#pragma unroll
            for (int kc = 0; kc < 4; ++kc)
#pragma unroll
                for (int g = 0; g < 4; ++g)
                    acc[g] = __builtin_amdgcn_wmma_f32_16x16x32_f16(
                        false, a0[kc], false, w0[g][kc], (short)0, acc[g], false, false);
        }
        __syncthreads();   // all reads of old h0 complete before overwrite

#pragma unroll
        for (int j = 0; j < 8; ++j) {
            const float ig = fast_sigm(acc[0][j]);
            const float fg = fast_sigm(acc[1][j]);
            const float gg = fast_tanh(acc[2][j]);
            const float og = fast_sigm(acc[3][j]);
            const float c  = fmaf(fg, c0[j], ig * gg);
            c0[j] = c;
            const float h = og * fast_tanh(c);
            lds_h0[(j + 8 * lh) * STR + 16 * w + li] = (_Float16)h;
        }
        __syncthreads();   // new h0 (layer-1 input) visible to all waves

        // ================= layer 1 =================
        // gates1 = h0_t @ Wih1^T + h1_{t-1} @ Whh1^T + b1
#pragma unroll
        for (int g = 0; g < 4; ++g)
#pragma unroll
            for (int j = 0; j < 8; ++j)
                acc[g][j] = bias1[g];

        // software-pipeline: issue next step's x loads under the layer-1 WMMAs
        const int tn = (t + 1 < SEQN) ? (t + 1) : t;
        float xnext[8];
#pragma unroll
        for (int j = 0; j < 8; ++j)
            xnext[j] = x[tn * BATCH + rb + j + 8 * lh];

        {
            v16h a1[8];
#pragma unroll
            for (int kc = 0; kc < 4; ++kc)
                a1[kc] = load_a_frag(lds_h0, 32 * kc, li, lh);
#pragma unroll
            for (int kc = 0; kc < 4; ++kc)
                a1[4 + kc] = load_a_frag(lds_h1, 32 * kc, li, lh);
#pragma unroll
            for (int kc = 0; kc < 4; ++kc)
#pragma unroll
                for (int g = 0; g < 4; ++g)
                    acc[g] = __builtin_amdgcn_wmma_f32_16x16x32_f16(
                        false, a1[kc], false, w1i[g][kc], (short)0, acc[g], false, false);
#pragma unroll
            for (int kc = 0; kc < 4; ++kc)
#pragma unroll
                for (int g = 0; g < 4; ++g)
                    acc[g] = __builtin_amdgcn_wmma_f32_16x16x32_f16(
                        false, a1[4 + kc], false, w1h[g][kc], (short)0, acc[g], false, false);
        }
        __syncthreads();   // all reads of old h1 complete before overwrite

#pragma unroll
        for (int j = 0; j < 8; ++j) {
            const float ig = fast_sigm(acc[0][j]);
            const float fg = fast_sigm(acc[1][j]);
            const float gg = fast_tanh(acc[2][j]);
            const float og = fast_sigm(acc[3][j]);
            const float c  = fmaf(fg, c1[j], ig * gg);
            c1[j] = c;
            const float h = og * fast_tanh(c);
            lds_h1[(j + 8 * lh) * STR + 16 * w + li] = (_Float16)h;
            xv[j] = xnext[j];   // commit pipelined x
        }
        __syncthreads();   // new h1 visible

        // Output: reshape(B,S,H)[-1] selects h2 of the last 8 timesteps.
        // out[(t-2040)*256 + b] = h2[t][b] . Wlin + blin. One writer per element.
        if (t >= SEQN - 8 && tid < ROWS) {
            float s = 0.f;
            for (int c = 0; c < HID; ++c)
                s += (float)lds_h1[tid * STR + c] * Wlin[c];
            out[(t - (SEQN - 8)) * BATCH + rb + tid] = s + blin[0];
        }
    }
}

extern "C" void kernel_launch(void* const* d_in, const int* in_sizes, int n_in,
                              void* d_out, int out_size, void* d_ws, size_t ws_size,
                              hipStream_t stream) {
    (void)in_sizes; (void)n_in; (void)d_ws; (void)ws_size; (void)out_size;
    lstm2_wmma_kernel<<<dim3(NWG), dim3(256), 0, stream>>>(
        (const float*)d_in[0],                          // x
        (const float*)d_in[1], (const float*)d_in[2],   // W_ih_l0, W_hh_l0
        (const float*)d_in[3], (const float*)d_in[4],   // b_ih_l0, b_hh_l0
        (const float*)d_in[5], (const float*)d_in[6],   // W_ih_l1, W_hh_l1
        (const float*)d_in[7], (const float*)d_in[8],   // b_ih_l1, b_hh_l1
        (const float*)d_in[9], (const float*)d_in[10],  // W_lin, b_lin
        (float*)d_out);
}
